// LstmEncoder_51754355917574
// MI455X (gfx1250) — compile-verified, matrix operand order checked by reading, not run on hardware
//
#include <hip/hip_runtime.h>

typedef __bf16 bf16_t;
typedef bf16_t bf16x8  __attribute__((ext_vector_type(8)));
typedef bf16_t bf16x16 __attribute__((ext_vector_type(16)));
typedef float  f32x8   __attribute__((ext_vector_type(8)));

#define TT 256   // timesteps
#define BB 256   // batch

// Load one 16x32 (A) or 32x16 (B) bf16 WMMA operand slice for this lane.
// Lane layout (wave32, 16-bit operands): element j <-> K = (j/8)*16 + hi*8 + (j%8),
// i.e. two contiguous 8-element (16B) chunks at +hi*8 and +16+hi*8.
__device__ __forceinline__ bf16x16 ld_tile(const bf16_t* p, int hi) {
  union { bf16x16 v; bf16x8 h[2]; } u;
  u.h[0] = *(const bf16x8*)(p + (hi << 3));
  u.h[1] = *(const bf16x8*)(p + 16 + (hi << 3));
  return u.v;
}

__device__ __forceinline__ float sigm(float x) { return 1.f / (1.f + __expf(-x)); }
__device__ __forceinline__ float tanh_(float x) {
  float xx = fminf(fmaxf(x, -15.f), 15.f);
  float e  = __expf(2.f * xx);
  return __fdividef(e - 1.f, e + 1.f);
}

// One LSTM layer scan. Each block owns 16 batch rows; wave w owns hidden cols
// [16w,16w+16) and computes all 4 gate tiles for them. h double-buffered in LDS,
// c stays in VGPRs (the WMMA C/D layout). Gates fused: K = D (input) + Hd (recurrent).
__global__ __launch_bounds__(512) void k_lstm_scan(
    const bf16_t* __restrict__ in,   // [T,B,D] bf16
    const bf16_t* __restrict__ wih,  // [4Hd,D] bf16 (row-major, PyTorch gate order i,f,g,o)
    const bf16_t* __restrict__ whh,  // [4Hd,Hd] bf16
    const float*  __restrict__ bias, // [4Hd] fp32
    bf16_t* __restrict__ out,        // [T,B,Hd] bf16
    float*  __restrict__ dout,       // [B,Hd] fp32 final hidden (nullptr if unused)
    int D, int Hd)
{
  __shared__ bf16_t hbuf[2][16 * 256];

  const int tid  = threadIdx.x;
  const int wave = tid >> 5;
  const int lane = tid & 31;
  const int l    = lane & 15;     // M row (A) / N col (B,C,D)
  const int hi   = lane >> 4;     // K half selector (A/B), M+8 selector (C/D)
  const int j0   = wave << 4;     // hidden-column base for this wave
  const int m0   = blockIdx.x << 4; // batch-row base for this block

  for (int i = tid; i < 16 * Hd; i += blockDim.x) hbuf[0][i] = (bf16_t)0.f;
  __syncthreads();

  const float bi  = bias[0 * Hd + j0 + l];
  const float bff = bias[1 * Hd + j0 + l];
  const float bg  = bias[2 * Hd + j0 + l];
  const float bo  = bias[3 * Hd + j0 + l];

  // B-operand row pointers: lane l supplies weight row (gate*Hd + j0 + l)
  const bf16_t* wi0 = wih + (size_t)(0 * Hd + j0 + l) * D;
  const bf16_t* wi1 = wih + (size_t)(1 * Hd + j0 + l) * D;
  const bf16_t* wi2 = wih + (size_t)(2 * Hd + j0 + l) * D;
  const bf16_t* wi3 = wih + (size_t)(3 * Hd + j0 + l) * D;
  const bf16_t* wh0 = whh + (size_t)(0 * Hd + j0 + l) * Hd;
  const bf16_t* wh1 = whh + (size_t)(1 * Hd + j0 + l) * Hd;
  const bf16_t* wh2 = whh + (size_t)(2 * Hd + j0 + l) * Hd;
  const bf16_t* wh3 = whh + (size_t)(3 * Hd + j0 + l) * Hd;

  f32x8 c = {0.f, 0.f, 0.f, 0.f, 0.f, 0.f, 0.f, 0.f};
  int cur = 0;

  for (int t = 0; t < TT; ++t) {
    f32x8 a0 = {0.f,0.f,0.f,0.f,0.f,0.f,0.f,0.f};
    f32x8 a1 = {0.f,0.f,0.f,0.f,0.f,0.f,0.f,0.f};
    f32x8 a2 = {0.f,0.f,0.f,0.f,0.f,0.f,0.f,0.f};
    f32x8 a3 = {0.f,0.f,0.f,0.f,0.f,0.f,0.f,0.f};

    const bf16_t* arow = in + ((size_t)t * BB + m0 + l) * D;
    if (t + 1 < TT)
      __builtin_prefetch(in + ((size_t)(t + 1) * BB + m0 + l) * D, 0, 0);

    // phase 1: input contribution, K over D
    for (int k0 = 0; k0 < D; k0 += 32) {
      bf16x16 av = ld_tile(arow + k0, hi);
      a0 = __builtin_amdgcn_wmma_f32_16x16x32_bf16(false, av, false, ld_tile(wi0 + k0, hi), (short)0, a0, false, false);
      a1 = __builtin_amdgcn_wmma_f32_16x16x32_bf16(false, av, false, ld_tile(wi1 + k0, hi), (short)0, a1, false, false);
      a2 = __builtin_amdgcn_wmma_f32_16x16x32_bf16(false, av, false, ld_tile(wi2 + k0, hi), (short)0, a2, false, false);
      a3 = __builtin_amdgcn_wmma_f32_16x16x32_bf16(false, av, false, ld_tile(wi3 + k0, hi), (short)0, a3, false, false);
    }
    // phase 2: recurrent contribution, K over Hd, A from LDS h
    const bf16_t* hrow = &hbuf[cur][l * Hd];
    for (int k0 = 0; k0 < Hd; k0 += 32) {
      bf16x16 av = ld_tile(hrow + k0, hi);
      a0 = __builtin_amdgcn_wmma_f32_16x16x32_bf16(false, av, false, ld_tile(wh0 + k0, hi), (short)0, a0, false, false);
      a1 = __builtin_amdgcn_wmma_f32_16x16x32_bf16(false, av, false, ld_tile(wh1 + k0, hi), (short)0, a1, false, false);
      a2 = __builtin_amdgcn_wmma_f32_16x16x32_bf16(false, av, false, ld_tile(wh2 + k0, hi), (short)0, a2, false, false);
      a3 = __builtin_amdgcn_wmma_f32_16x16x32_bf16(false, av, false, ld_tile(wh3 + k0, hi), (short)0, a3, false, false);
    }

    const int nxt = cur ^ 1;
    const size_t ob = ((size_t)t * BB + m0) * (size_t)Hd + j0 + l;
#pragma unroll
    for (int r = 0; r < 8; ++r) {
      const int m = r + (hi << 3);          // C/D row in tile
      float iv = sigm (a0[r] + bi);
      float fv = sigm (a1[r] + bff);
      float gv = tanh_(a2[r] + bg);
      float ov = sigm (a3[r] + bo);
      float cc = fv * c[r] + iv * gv;
      c[r] = cc;
      float hv = ov * tanh_(cc);
      bf16_t hb = (bf16_t)hv;
      hbuf[nxt][m * Hd + j0 + l] = hb;
      out[ob + (size_t)m * Hd] = hb;
      if (dout && t == TT - 1) dout[(m0 + m) * Hd + j0 + l] = hv;
    }
    __syncthreads();   // h[nxt] complete & visible before next step reads it
    cur = nxt;
  }
}

// fp32 -> bf16 flat convert
__global__ void k_cvt_bf16(const float* __restrict__ in, bf16_t* __restrict__ out, int n) {
  int i = blockIdx.x * blockDim.x + threadIdx.x;
  if (i < n) out[i] = (bf16_t)in[i];
}

// x [B,T,8] fp32 -> xb [T,B,32] bf16, zero-padded K (8 -> 32)
__global__ void k_pad_x(const float* __restrict__ x, bf16_t* __restrict__ xb) {
  int i = blockIdx.x * blockDim.x + threadIdx.x;
  if (i >= TT * BB * 32) return;
  int k = i & 31;
  int tb = i >> 5;
  int b = tb & (BB - 1);
  int t = tb >> 8;
  xb[i] = (k < 8) ? (bf16_t)x[((size_t)b * TT + t) * 8 + k] : (bf16_t)0.f;
}

// w_ih0_1 [1024,8] fp32 -> [1024,32] bf16, zero-padded
__global__ void k_pad_w(const float* __restrict__ w, bf16_t* __restrict__ wp) {
  int i = blockIdx.x * blockDim.x + threadIdx.x;
  if (i >= 1024 * 32) return;
  int k = i & 31;
  int r = i >> 5;
  wp[i] = (k < 8) ? (bf16_t)w[r * 8 + k] : (bf16_t)0.f;
}

extern "C" void kernel_launch(void* const* d_in, const int* in_sizes, int n_in,
                              void* d_out, int out_size, void* d_ws, size_t ws_size,
                              hipStream_t stream) {
  (void)in_sizes; (void)n_in; (void)out_size; (void)ws_size;

  const float* x     = (const float*)d_in[0];
  const float* wih01 = (const float*)d_in[1];
  const float* wihr1 = (const float*)d_in[2];
  const float* whh1  = (const float*)d_in[3];
  const float* b1    = (const float*)d_in[4];
  const float* wih02 = (const float*)d_in[5];
  const float* wihr2 = (const float*)d_in[6];
  const float* whh2  = (const float*)d_in[7];
  const float* b2    = (const float*)d_in[8];

  char* p = (char*)d_ws;
  auto carve = [&](size_t bytes) { char* r = p; p += (bytes + 255) & ~(size_t)255; return r; };
  bf16_t* xb     = (bf16_t*)carve((size_t)TT * BB * 32 * 2);      // 4 MB
  bf16_t* wp0    = (bf16_t*)carve((size_t)1024 * 32 * 2);
  bf16_t* cwihr1 = (bf16_t*)carve((size_t)4 * 1024 * 256 * 2);
  bf16_t* cwhh1  = (bf16_t*)carve((size_t)5 * 1024 * 256 * 2);
  bf16_t* cwih02 = (bf16_t*)carve((size_t)512 * 256 * 2);
  bf16_t* cwihr2 = (bf16_t*)carve((size_t)4 * 512 * 128 * 2);
  bf16_t* cwhh2  = (bf16_t*)carve((size_t)5 * 512 * 128 * 2);
  bf16_t* yA     = (bf16_t*)carve((size_t)TT * BB * 256 * 2);     // 32 MB
  bf16_t* yB     = (bf16_t*)carve((size_t)TT * BB * 256 * 2);     // 32 MB

  auto cvt = [&](const float* src, bf16_t* dst, int n) {
    k_cvt_bf16<<<(n + 255) / 256, 256, 0, stream>>>(src, dst, n);
  };
  k_pad_x<<<(TT * BB * 32 + 255) / 256, 256, 0, stream>>>(x, xb);
  k_pad_w<<<(1024 * 32 + 255) / 256, 256, 0, stream>>>(wih01, wp0);
  cvt(wihr1, cwihr1, 4 * 1024 * 256);
  cvt(whh1,  cwhh1,  5 * 1024 * 256);
  cvt(wih02, cwih02, 512 * 256);
  cvt(wihr2, cwihr2, 4 * 512 * 128);
  cvt(whh2,  cwhh2,  5 * 512 * 128);

  bf16_t* bufs[2] = {yA, yB};
  int ping = 0;

  // ---- stack 1: D in {32(pad), 256}, Hd = 256, 16 waves/block ----
  k_lstm_scan<<<16, 512, 0, stream>>>(xb, wp0, cwhh1, b1, bufs[0], nullptr, 32, 256);
  for (int layer = 1; layer < 5; ++layer) {
    k_lstm_scan<<<16, 512, 0, stream>>>(
        bufs[ping],
        cwihr1 + (size_t)(layer - 1) * 1024 * 256,
        cwhh1  + (size_t)layer * 1024 * 256,
        b1 + layer * 1024,
        bufs[ping ^ 1], nullptr, 256, 256);
    ping ^= 1;
  }

  // ---- stack 2: D in {256, 128}, Hd = 128, 8 waves/block ----
  k_lstm_scan<<<16, 256, 0, stream>>>(bufs[ping], cwih02, cwhh2, b2,
                                      bufs[ping ^ 1], nullptr, 256, 128);
  ping ^= 1;
  for (int layer = 1; layer < 5; ++layer) {
    k_lstm_scan<<<16, 256, 0, stream>>>(
        bufs[ping],
        cwihr2 + (size_t)(layer - 1) * 512 * 128,
        cwhh2  + (size_t)layer * 512 * 128,
        b2 + layer * 512,
        bufs[ping ^ 1],
        (layer == 4) ? (float*)d_out : nullptr, 128, 128);
    ping ^= 1;
  }
}
